// GnnLayerwithVirtualNode_32014686224547
// MI455X (gfx1250) — matrix operand order repaired and checked by reading, not compiled, not run on hardware
//
#include <hip/hip_runtime.h>

typedef __attribute__((ext_vector_type(16))) _Float16 v16h;
typedef __attribute__((ext_vector_type(8)))  _Float16 v8h;
typedef __attribute__((ext_vector_type(8)))  float    v8f;

#define EMB 128
#define HID 256
#define NGRAPH 512
#define NLAYER 3
#define BN_INV 0.9999950000374997f   // 1/sqrt(1+1e-5)

union V16H { v16h v; _Float16 e[16]; };

__device__ __forceinline__ void atomic_add_f32(float* p, float v) {
    __hip_atomic_fetch_add(p, v, __ATOMIC_RELAXED, __HIP_MEMORY_SCOPE_AGENT);
}

// ---------------------------------------------------------------------------
// Pack a row-major K x NC f32 weight matrix into WMMA-B f16 fragments.
// Fragment f = kb*NT + nt; lane holds 16 halves: B[kb*32 + (lane>=16?16:0) + j][nt*16 + (lane&15)]
// ---------------------------------------------------------------------------
__global__ void pack_b_kernel(const float* __restrict__ W, _Float16* __restrict__ dst,
                              int K, int NC) {
    int NT = NC >> 4;
    int f = blockIdx.x;                // kb*NT + nt
    int kb = f / NT, nt = f % NT;
    int lane = threadIdx.x;            // 0..31
    int kbase = kb * 32 + ((lane & 16) ? 16 : 0);
    int col = nt * 16 + (lane & 15);
    _Float16* o = dst + ((size_t)f * 32 + lane) * 16;
#pragma unroll
    for (int j = 0; j < 16; ++j)
        o[j] = (_Float16)W[(size_t)(kbase + j) * NC + col];
}

__global__ void vn_init_kernel(const float* __restrict__ ve, float* __restrict__ vn) {
    int i = blockIdx.x * blockDim.x + threadIdx.x;   // NGRAPH*EMB threads
    vn[i] = ve[i & (EMB - 1)];
}

// h_in = h + vn[batch];  z-accumulator initialized to h_in (so z becomes h_in + agg)
__global__ void hin_kernel(const float* __restrict__ h, const float* __restrict__ vn,
                           const long long* __restrict__ batch,
                           float* __restrict__ hin, float* __restrict__ z, int N) {
    int t = blockIdx.x * blockDim.x + threadIdx.x;
    int node = t >> 5;
    if (node >= N) return;
    int c = (t & 31) * 4;
    long long g = batch[node];
    const float4 x = *(const float4*)(h  + (size_t)node * EMB + c);
    const float4 v = *(const float4*)(vn + (size_t)g    * EMB + c);
    float4 s; s.x = x.x + v.x; s.y = x.y + v.y; s.z = x.z + v.z; s.w = x.w + v.w;
    *(float4*)(hin + (size_t)node * EMB + c) = s;
    *(float4*)(z   + (size_t)node * EMB + c) = s;
}

// One wave per edge: emb = edge_attr @ We + be (We in LDS), msg = relu(h_in[src]+emb),
// scatter-add into z[dst]. Lane owns 4 of the 128 columns.
__global__ void edge_kernel(const float* __restrict__ hin, const float* __restrict__ eattr,
                            const long long* __restrict__ eidx, const float* __restrict__ We,
                            const float* __restrict__ be, float* __restrict__ z,
                            int E, int nwaves) {
    __shared__ float sW[16 * EMB];
    __shared__ float sB[EMB];
    for (int i = threadIdx.x; i < 16 * EMB; i += blockDim.x) sW[i] = We[i];
    for (int i = threadIdx.x; i < EMB; i += blockDim.x)      sB[i] = be[i];
    __syncthreads();
    int lane = threadIdx.x & 31;
    int wave = (blockIdx.x * blockDim.x + threadIdx.x) >> 5;
    int c = lane * 4;
    for (int e = wave; e < E; e += nwaves) {
        long long s = eidx[e];
        long long d = eidx[(size_t)E + e];
        float a0 = sB[c], a1 = sB[c + 1], a2 = sB[c + 2], a3 = sB[c + 3];
#pragma unroll
        for (int i = 0; i < 16; ++i) {
            float a = eattr[(size_t)e * 16 + i];            // wave-uniform broadcast
            const float4 w = *(const float4*)(&sW[i * EMB + c]);
            a0 += a * w.x; a1 += a * w.y; a2 += a * w.z; a3 += a * w.w;
        }
        const float4 x = *(const float4*)(hin + (size_t)s * EMB + c);
        float m0 = fmaxf(x.x + a0, 0.f);
        float m1 = fmaxf(x.y + a1, 0.f);
        float m2 = fmaxf(x.z + a2, 0.f);
        float m3 = fmaxf(x.w + a3, 0.f);
        float* zp = z + (size_t)d * EMB + c;
        atomic_add_f32(zp + 0, m0);
        atomic_add_f32(zp + 1, m1);
        atomic_add_f32(zp + 2, m2);
        atomic_add_f32(zp + 3, m3);
    }
}

// WMMA node MLP: out = BN(relu(z@W1+b1)@W2+b2) [relu] + h_in.  One wave = 16 nodes.
__global__ void __launch_bounds__(128) mlp_kernel(
        const float* __restrict__ z, const float* __restrict__ hin,
        const _Float16* __restrict__ pw1, const _Float16* __restrict__ pw2,
        const float* __restrict__ b1, const float* __restrict__ b2,
        const float* __restrict__ gamma, const float* __restrict__ beta,
        float* __restrict__ out, int N, int relu_out) {
    __shared__ __align__(16) _Float16 H[4][16 * HID];
    int w = threadIdx.x >> 5, lane = threadIdx.x & 31;
    int tile = blockIdx.x * 4 + w;
    if (tile * 16 >= N) return;
    int mrow = lane & 15;
    int node = tile * 16 + mrow;
    if (node >= N) node = N - 1;
    int koff  = (lane & 16) ? 16 : 0;
    int rbase = (lane & 16) ? 8 : 0;
    int colq  = lane & 15;

    // A fragments for GEMM1 (K = 128 -> 4 k-blocks), f32 -> f16
    V16H a1[4];
    const float* row = z + (size_t)node * EMB;
#pragma unroll
    for (int kb = 0; kb < 4; ++kb) {
        int k0 = kb * 32 + koff;
        float4 f0 = *(const float4*)(row + k0);
        float4 f1 = *(const float4*)(row + k0 + 4);
        float4 f2 = *(const float4*)(row + k0 + 8);
        float4 f3 = *(const float4*)(row + k0 + 12);
        a1[kb].e[0]  = (_Float16)f0.x; a1[kb].e[1]  = (_Float16)f0.y;
        a1[kb].e[2]  = (_Float16)f0.z; a1[kb].e[3]  = (_Float16)f0.w;
        a1[kb].e[4]  = (_Float16)f1.x; a1[kb].e[5]  = (_Float16)f1.y;
        a1[kb].e[6]  = (_Float16)f1.z; a1[kb].e[7]  = (_Float16)f1.w;
        a1[kb].e[8]  = (_Float16)f2.x; a1[kb].e[9]  = (_Float16)f2.y;
        a1[kb].e[10] = (_Float16)f2.z; a1[kb].e[11] = (_Float16)f2.w;
        a1[kb].e[12] = (_Float16)f3.x; a1[kb].e[13] = (_Float16)f3.y;
        a1[kb].e[14] = (_Float16)f3.z; a1[kb].e[15] = (_Float16)f3.w;
    }

    _Float16* Hw = H[w];
    // GEMM1: 16 n-tiles of the 256-wide hidden layer
#pragma unroll
    for (int nt = 0; nt < 16; ++nt) {
        v8f acc = {};
#pragma unroll
        for (int kb = 0; kb < 4; ++kb) {
            v16h b = *(const v16h*)(pw1 + ((size_t)(kb * 16 + nt) * 32 + lane) * 16);
            acc = __builtin_amdgcn_wmma_f32_16x16x32_f16(false, a1[kb].v, false, b,
                                                         (short)0, acc, false, false);
        }
        int col = nt * 16 + colq;
        float bias = b1[col];
#pragma unroll
        for (int r = 0; r < 8; ++r) {
            float v = fmaxf(acc[r] + bias, 0.f);
            Hw[(rbase + r) * HID + col] = (_Float16)v;
        }
    }

    // GEMM2: K = 256 -> 8 k-blocks, 8 output n-tiles of 128
#pragma unroll
    for (int nt = 0; nt < 8; ++nt) {
        v8f acc = {};
#pragma unroll
        for (int kb = 0; kb < 8; ++kb) {
            const _Float16* ap = &Hw[mrow * HID + kb * 32 + koff];
            v8h lo = *(const v8h*)(ap);
            v8h hi = *(const v8h*)(ap + 8);
            v16h a2 = __builtin_shufflevector(lo, hi, 0,1,2,3,4,5,6,7,8,9,10,11,12,13,14,15);
            v16h b = *(const v16h*)(pw2 + ((size_t)(kb * 8 + nt) * 32 + lane) * 16);
            acc = __builtin_amdgcn_wmma_f32_16x16x32_f16(false, a2, false, b,
                                                         (short)0, acc, false, false);
        }
        int col = nt * 16 + colq;
        float bb = b2[col];
        float gs = gamma[col] * BN_INV;
        float bt = beta[col];
#pragma unroll
        for (int r = 0; r < 8; ++r) {
            int m = tile * 16 + rbase + r;
            int mc = (m < N) ? m : 0;
            float v = acc[r] + bb;
            v = v * gs + bt;
            if (relu_out) v = fmaxf(v, 0.f);
            v += hin[(size_t)mc * EMB + col];
            if (m < N) out[(size_t)m * EMB + col] = v;
        }
    }
}

__global__ void vt_init_kernel(const float* __restrict__ vn, float* __restrict__ vt) {
    int i = blockIdx.x * blockDim.x + threadIdx.x;
    vt[i] = vn[i];
}

__global__ void vt_acc_kernel(const float* __restrict__ hin, const long long* __restrict__ batch,
                              float* __restrict__ vt, int N) {
    int t = blockIdx.x * blockDim.x + threadIdx.x;
    int node = t >> 5;
    if (node >= N) return;
    int c = (t & 31) * 4;
    long long g = batch[node];
    const float4 x = *(const float4*)(hin + (size_t)node * EMB + c);
    float* p = vt + (size_t)g * EMB + c;
    atomic_add_f32(p + 0, x.x);
    atomic_add_f32(p + 1, x.y);
    atomic_add_f32(p + 2, x.z);
    atomic_add_f32(p + 3, x.w);
}

// Tiny virtual-node MLP: vn += relu(BN(relu(BN(vt@W1+b1))@W2+b2)).  One block per graph.
__global__ void __launch_bounds__(128) vnmlp_kernel(
        const float* __restrict__ vt, float* __restrict__ vn,
        const float* __restrict__ w1, const float* __restrict__ bb1,
        const float* __restrict__ g1, const float* __restrict__ be1,
        const float* __restrict__ w2, const float* __restrict__ bb2,
        const float* __restrict__ g2, const float* __restrict__ be2) {
    __shared__ float xr[EMB];
    __shared__ float mr[HID];
    int g = blockIdx.x, t = threadIdx.x;
    xr[t] = vt[(size_t)g * EMB + t];
    __syncthreads();
#pragma unroll
    for (int cc = 0; cc < 2; ++cc) {
        int c = t + cc * 128;
        float s = bb1[c];
        for (int k = 0; k < EMB; ++k) s += xr[k] * w1[(size_t)k * HID + c];
        s = s * (g1[c] * BN_INV) + be1[c];
        mr[c] = fmaxf(s, 0.f);
    }
    __syncthreads();
    float s = bb2[t];
    for (int k = 0; k < HID; ++k) s += mr[k] * w2[(size_t)k * EMB + t];
    s = s * (g2[t] * BN_INV) + be2[t];
    s = fmaxf(s, 0.f);
    vn[(size_t)g * EMB + t] += s;
}

// ---------------------------------------------------------------------------
extern "C" void kernel_launch(void* const* d_in, const int* in_sizes, int n_in,
                              void* d_out, int out_size, void* d_ws, size_t ws_size,
                              hipStream_t stream) {
    const float*     x      = (const float*)d_in[0];
    const float*     eattr  = (const float*)d_in[1];
    const long long* eidx   = (const long long*)d_in[2];
    const long long* batch  = (const long long*)d_in[3];
    const float*     vn_emb = (const float*)d_in[4];
    const float*     cew    = (const float*)d_in[5];
    const float*     ceb    = (const float*)d_in[6];
    const float*     cw1    = (const float*)d_in[7];
    const float*     cb1    = (const float*)d_in[8];
    const float*     cw2    = (const float*)d_in[9];
    const float*     cb2    = (const float*)d_in[10];
    const float*     bng    = (const float*)d_in[11];
    const float*     bnb    = (const float*)d_in[12];
    const float*     vw1    = (const float*)d_in[13];
    const float*     vb1    = (const float*)d_in[14];
    const float*     vg1    = (const float*)d_in[15];
    const float*     vbe1   = (const float*)d_in[16];
    const float*     vw2    = (const float*)d_in[17];
    const float*     vb2    = (const float*)d_in[18];
    const float*     vg2    = (const float*)d_in[19];
    const float*     vbe2   = (const float*)d_in[20];

    int N = in_sizes[0] / EMB;
    int E = in_sizes[1] / 16;

    char* ws = (char*)d_ws;
    size_t nodeBytes = (size_t)N * EMB * sizeof(float);
    float* buf0 = (float*)ws; ws += nodeBytes;                       // h ping buffer
    float* hin  = (float*)ws; ws += nodeBytes;                       // h + vn[batch]
    float* zbuf = (float*)ws; ws += nodeBytes;                       // h_in + agg
    float* vn   = (float*)ws; ws += (size_t)NGRAPH * EMB * sizeof(float);
    float* vt   = (float*)ws; ws += (size_t)NGRAPH * EMB * sizeof(float);
    _Float16* pw1 = (_Float16*)ws; ws += (size_t)NLAYER * EMB * HID * sizeof(_Float16);
    _Float16* pw2 = (_Float16*)ws; ws += (size_t)NLAYER * HID * EMB * sizeof(_Float16);

    // One-time prep (re-done every call for determinism): vn table + packed weights
    vn_init_kernel<<<(NGRAPH * EMB) / 256, 256, 0, stream>>>(vn_emb, vn);
    for (int l = 0; l < NLAYER; ++l) {
        pack_b_kernel<<<(EMB / 32) * (HID / 16), 32, 0, stream>>>(
            cw1 + (size_t)l * EMB * HID, pw1 + (size_t)l * EMB * HID, EMB, HID);
        pack_b_kernel<<<(HID / 32) * (EMB / 16), 32, 0, stream>>>(
            cw2 + (size_t)l * HID * EMB, pw2 + (size_t)l * HID * EMB, HID, EMB);
    }

    int nodeThreads = N * 32;
    int hinBlocks = (nodeThreads + 255) / 256;
    int tiles = (N + 15) / 16;
    int mlpBlocks = (tiles + 3) / 4;
    int edgeBlocks = 1024;
    int nwaves = edgeBlocks * 8;

    const float* hcur = x;
    for (int l = 0; l < NLAYER; ++l) {
        hin_kernel<<<hinBlocks, 256, 0, stream>>>(hcur, vn, batch, hin, zbuf, N);
        edge_kernel<<<edgeBlocks, 256, 0, stream>>>(
            hin, eattr, eidx, cew + (size_t)l * 16 * EMB, ceb + (size_t)l * EMB,
            zbuf, E, nwaves);
        float* outp = (l == NLAYER - 1) ? (float*)d_out : buf0;
        mlp_kernel<<<mlpBlocks, 128, 0, stream>>>(
            zbuf, hin,
            pw1 + (size_t)l * EMB * HID, pw2 + (size_t)l * HID * EMB,
            cb1 + (size_t)l * HID, cb2 + (size_t)l * EMB,
            bng + (size_t)l * EMB, bnb + (size_t)l * EMB,
            outp, N, (l < NLAYER - 1) ? 1 : 0);
        if (l < NLAYER - 1) {
            vt_init_kernel<<<(NGRAPH * EMB) / 256, 256, 0, stream>>>(vn, vt);
            vt_acc_kernel<<<hinBlocks, 256, 0, stream>>>(hin, batch, vt, N);
            vnmlp_kernel<<<NGRAPH, 128, 0, stream>>>(
                vt, vn,
                vw1 + (size_t)l * EMB * HID, vb1 + (size_t)l * HID,
                vg1 + (size_t)l * HID,       vbe1 + (size_t)l * HID,
                vw2 + (size_t)l * HID * EMB, vb2 + (size_t)l * EMB,
                vg2 + (size_t)l * EMB,       vbe2 + (size_t)l * EMB);
        }
        hcur = buf0;
    }
}
